// FullAttention_9766755631492
// MI455X (gfx1250) — matrix operand order repaired and checked by reading, not compiled, not run on hardware
//
#include <hip/hip_runtime.h>

// ---------------------------------------------------------------------------
// Causal attention forward (returns V and full probability matrix A).
// B=4, L=2048, H=16, E=64, fp32 in/out.  Two-pass softmax:
//   pass 1: flash-style running (rowmax, rowsumexp) via bf16 WMMA QK^T
//   pass 2: recompute S, P = exp(scale*S - m)/s, write A once (non-temporal),
//           accumulate O += P*V via bf16 WMMA.
// Roofline: mandatory traffic ~1.13 GB (1 GB A write dominant) -> ~49 us at
// 23.3 TB/s; bf16 WMMA compute (~50 GFLOP w/ causal skipping) is negligible,
// so the kernel streams A with non-temporal stores and prefetches K/V tiles.
// ---------------------------------------------------------------------------

typedef __bf16 bf16_t;
typedef bf16_t v16bf __attribute__((ext_vector_type(16)));
typedef float  v8f   __attribute__((ext_vector_type(8)));
typedef float  v4f   __attribute__((ext_vector_type(4)));

#define DI __device__ __forceinline__

static constexpr int B_ = 4, L_ = 2048, H_ = 16, E_ = 64;
static constexpr int ROWS_WG = 128;           // rows per workgroup (8 waves x 16)
static constexpr int CT = 64;                 // key/value column tile
static constexpr int NT = L_ / CT;            // 32 column tiles
static constexpr int SL = 3 * H_ * E_;        // 3072: qkv row stride (floats)
static constexpr int KS = 72, VS = 72, PS = 72; // padded LDS strides (elements)
static constexpr float SCALE = 0.125f;        // 1/sqrt(64)

union Frag { v16bf v; uint4 q[2]; unsigned int u[8]; };

DI unsigned short f2bf(float f) {
  unsigned int u = __builtin_bit_cast(unsigned int, f);
  u += 0x7FFFu + ((u >> 16) & 1u);            // round-to-nearest-even
  return (unsigned short)(u >> 16);
}
DI unsigned int pk2(float a, float b) {
  return (unsigned int)f2bf(a) | ((unsigned int)f2bf(b) << 16);
}

DI v8f wmma_bf16(v16bf a, v16bf b, v8f c) {
  // D(16x16,f32) = A(16x32,bf16) * B(32x16,bf16) + C
  return __builtin_amdgcn_wmma_f32_16x16x32_bf16(false, a, false, b,
                                                 (short)0, c, false, false);
}

// --- fragment builders (ISA 7.12.2 layouts, wave32) ------------------------

// A-operand (Q, 16x32 bf16) loaded straight from global fp32 rows.
DI v16bf load_aQ(const float* qrow, int lane, int e0) {
  const int m  = lane & 15;
  const int kb = (lane < 16) ? 0 : 8;
  const float* r = qrow + (size_t)m * SL;
  Frag f;
  const v4f a0 = *(const v4f*)(r + e0 + kb);
  const v4f a1 = *(const v4f*)(r + e0 + kb + 4);
  const v4f b0 = *(const v4f*)(r + e0 + 16 + kb);
  const v4f b1 = *(const v4f*)(r + e0 + 16 + kb + 4);
  f.u[0] = pk2(a0.x, a0.y); f.u[1] = pk2(a0.z, a0.w);
  f.u[2] = pk2(a1.x, a1.y); f.u[3] = pk2(a1.z, a1.w);
  f.u[4] = pk2(b0.x, b0.y); f.u[5] = pk2(b0.z, b0.w);
  f.u[6] = pk2(b1.x, b1.y); f.u[7] = pk2(b1.z, b1.w);
  return f.v;
}

// B-operand (K^T chunk, 32x16 bf16) from LDS K tile stored [s][e].
DI v16bf load_bK(const unsigned short* Ksm, int lane, int n, int e0) {
  const int j  = lane & 15;
  const int kb = (lane < 16) ? 0 : 8;
  const unsigned short* p = &Ksm[(n * 16 + j) * KS + e0 + kb];
  Frag f;
  f.q[0] = *(const uint4*)(p);
  f.q[1] = *(const uint4*)(p + 16);
  return f.v;
}

// B-operand (V chunk, 32x16 bf16) from LDS V tile stored transposed [e][s].
DI v16bf load_bV(const unsigned short* Vsm, int lane, int n2, int kc) {
  const int j  = lane & 15;
  const int kb = (lane < 16) ? 0 : 8;
  const unsigned short* p = &Vsm[(n2 * 16 + j) * VS + kc + kb];
  Frag f;
  f.q[0] = *(const uint4*)(p);
  f.q[1] = *(const uint4*)(p + 16);
  return f.v;
}

// A-operand (P, 16x32 bf16) from wave-private LDS P tile stored [m][s].
DI v16bf load_aP(const unsigned short* Pw, int lane, int k0) {
  const int m  = lane & 15;
  const int kb = (lane < 16) ? 0 : 8;
  const unsigned short* p = &Pw[m * PS + k0 + kb];
  Frag f;
  f.q[0] = *(const uint4*)(p);
  f.q[1] = *(const uint4*)(p + 16);
  return f.v;
}

// Cooperative stage of one 64x64 K (and optionally V) tile into LDS (bf16).
DI void stage_kv(const float* kbase, const float* vbase,
                 unsigned short* Ksm, unsigned short* Vsm,
                 int tid, bool loadV) {
#pragma unroll
  for (int q = 0; q < 4; ++q) {
    const int li = tid + 256 * q;             // 1024 float4 units
    const int s  = li >> 4;                   // key/value row in tile
    const int e  = (li & 15) << 2;            // feature offset
    const v4f kf = *(const v4f*)(kbase + (size_t)s * SL + e);
    *(uint2*)(&Ksm[s * KS + e]) = make_uint2(pk2(kf.x, kf.y), pk2(kf.z, kf.w));
    if (loadV) {
      const v4f vf = *(const v4f*)(vbase + (size_t)s * SL + e);
      Vsm[(e + 0) * VS + s] = f2bf(vf.x);
      Vsm[(e + 1) * VS + s] = f2bf(vf.y);
      Vsm[(e + 2) * VS + s] = f2bf(vf.z);
      Vsm[(e + 3) * VS + s] = f2bf(vf.w);
    }
  }
}

// Prefetch one 64B cacheline per thread of the NEXT tile (256 lines = whole
// 64x64 f32 tile).  Lowers to global_prefetch_b8 (probe-confirmed).
DI void prefetch_tile(const float* base, int tid) {
  const int s = tid >> 2;                     // row in tile
  const int e = (tid & 3) << 4;               // 16 floats = one 64B line
  __builtin_prefetch(base + (size_t)s * SL + e, 0, 1);
}

__global__ __launch_bounds__(256)
void attn_fwd_cdna5(const float* __restrict__ qkv,
                    float* __restrict__ vout,
                    float* __restrict__ aout) {
  __shared__ __attribute__((aligned(16))) unsigned short Ksm[CT * KS];
  __shared__ __attribute__((aligned(16))) unsigned short Vsm[E_ * VS];
  __shared__ __attribute__((aligned(16))) unsigned short Psm[8 * 16 * PS];

  const int tid  = threadIdx.x;
  const int lane = tid & 31;
  const int wave = tid >> 5;
  const int hl   = lane >> 4;

  const int tileIdx = blockIdx.x & 15;        // 16 row tiles per (b,h)
  const int h       = (blockIdx.x >> 4) & 15;
  const int b       = blockIdx.x >> 8;

  const int rbase = tileIdx * ROWS_WG + wave * 16;   // this wave's 16 rows

  const size_t SB = (size_t)L_ * SL;
  const float* qw  = qkv + (size_t)b * SB + (size_t)rbase * SL + h * E_;      // Q
  const float* kb_ = qkv + (size_t)b * SB + 1 * H_ * E_ + h * E_;             // K
  const float* vb_ = qkv + (size_t)b * SB + 2 * H_ * E_ + h * E_;             // V
  float* Abh = aout + (size_t)(b * H_ + h) * L_ * L_;
  unsigned short* Pw = &Psm[wave * 16 * PS];

  // Q fragments live in registers for the whole kernel (E=64 -> two K-chunks).
  const v16bf aq0 = load_aQ(qw, lane, 0);
  const v16bf aq1 = load_aQ(qw, lane, 32);

  const float NINF = -__builtin_inff();
  float mrow[8], srow[8];
#pragma unroll
  for (int i = 0; i < 8; ++i) { mrow[i] = NINF; srow[i] = 0.0f; }

  const int ntc    = 2 * tileIdx + 2;         // col tiles touching this block
  const int mylast = (rbase + 15) >> 6;       // last col tile this wave needs

  // ---------------- Pass 1: row max and sum-of-exp -------------------------
  for (int ct = 0; ct < ntc; ++ct) {
    __syncthreads();
    stage_kv(kb_ + (size_t)ct * CT * SL, vb_, Ksm, Vsm, tid, false);
    __syncthreads();
    if (ct + 1 < ntc)                          // overlap next tile's latency
      prefetch_tile(kb_ + (size_t)(ct + 1) * CT * SL, tid);
    if (ct > mylast) continue;
    const int c0 = ct * CT;

    float sv[4][8];
#pragma unroll
    for (int n = 0; n < 4; ++n) {
      v8f acc = {};
      acc = wmma_bf16(aq0, load_bK(Ksm, lane, n, 0), acc);
      acc = wmma_bf16(aq1, load_bK(Ksm, lane, n, 32), acc);
      const int col = c0 + n * 16 + (lane & 15);
#pragma unroll
      for (int i = 0; i < 8; ++i) {
        const int row = rbase + i + 8 * hl;
        sv[n][i] = (col <= row) ? acc[i] * SCALE : NINF;
      }
    }
#pragma unroll
    for (int i = 0; i < 8; ++i) {
      float lm = fmaxf(fmaxf(sv[0][i], sv[1][i]), fmaxf(sv[2][i], sv[3][i]));
#pragma unroll
      for (int off = 8; off >= 1; off >>= 1)
        lm = fmaxf(lm, __shfl_xor(lm, off, 32));      // 16-lane half reduce
      const float nm = fmaxf(mrow[i], lm);
      float ls = __expf(sv[0][i] - nm) + __expf(sv[1][i] - nm) +
                 __expf(sv[2][i] - nm) + __expf(sv[3][i] - nm);
#pragma unroll
      for (int off = 8; off >= 1; off >>= 1)
        ls += __shfl_xor(ls, off, 32);
      srow[i] = srow[i] * __expf(mrow[i] - nm) + ls;
      mrow[i] = nm;
    }
  }

  float rs[8];
#pragma unroll
  for (int i = 0; i < 8; ++i) rs[i] = 1.0f / srow[i];

  // ---------------- Pass 2: write A once, O += P*V -------------------------
  v8f o0 = {}, o1 = {}, o2 = {}, o3 = {};
  for (int ct = 0; ct < NT; ++ct) {
    const int c0 = ct * CT;
    if (ct < ntc) {                            // block-uniform: barriers legal
      __syncthreads();
      stage_kv(kb_ + (size_t)ct * CT * SL, vb_ + (size_t)ct * CT * SL,
               Ksm, Vsm, tid, true);
      __syncthreads();
      if (ct + 1 < ntc) {                      // prefetch next K and V tiles
        prefetch_tile(kb_ + (size_t)(ct + 1) * CT * SL, tid);
        prefetch_tile(vb_ + (size_t)(ct + 1) * CT * SL, tid);
      }
    }
    if (ct > mylast) {
      // strictly above the diagonal for this wave: A tile is exactly zero
      const v4f z = {};
#pragma unroll
      for (int q = 0; q < 8; ++q) {
        const int fi = lane + 32 * q;          // 256 float4 per 16x64 tile
        const int r  = fi >> 4, c4 = (fi & 15) << 2;
        __builtin_nontemporal_store(
            z, (v4f*)(Abh + (size_t)(rbase + r) * L_ + c0 + c4));
      }
      continue;
    }

#pragma unroll
    for (int n = 0; n < 4; ++n) {
      v8f acc = {};
      acc = wmma_bf16(aq0, load_bK(Ksm, lane, n, 0), acc);
      acc = wmma_bf16(aq1, load_bK(Ksm, lane, n, 32), acc);
      const int col16 = n * 16 + (lane & 15);
      const int col   = c0 + col16;
#pragma unroll
      for (int i = 0; i < 8; ++i) {
        const int rloc = i + 8 * hl;
        const int row  = rbase + rloc;
        const float val = (col <= row) ? acc[i] * SCALE : NINF;
        const float p   = __expf(val - mrow[i]) * rs[i];   // masked -> 0
        __builtin_nontemporal_store(p, Abh + (size_t)row * L_ + col);
        Pw[rloc * PS + col16] = f2bf(p);
      }
    }
    // O(16x64) += P(16x64) * V(64x64): K-chunks {0,32}, four 16-wide N tiles
#pragma unroll
    for (int kc = 0; kc < 2; ++kc) {
      const v16bf ap = load_aP(Pw, lane, kc * 32);
      o0 = wmma_bf16(ap, load_bV(Vsm, lane, 0, kc * 32), o0);
      o1 = wmma_bf16(ap, load_bV(Vsm, lane, 1, kc * 32), o1);
      o2 = wmma_bf16(ap, load_bV(Vsm, lane, 2, kc * 32), o2);
      o3 = wmma_bf16(ap, load_bV(Vsm, lane, 3, kc * 32), o3);
    }
  }

  // ---------------- write O -> V output (B, L, H, E) -----------------------
#pragma unroll
  for (int i = 0; i < 8; ++i) {
    const int row = rbase + i + 8 * hl;
    float* vr = vout + ((size_t)(b * L_ + row) * H_ + h) * E_ + (lane & 15);
    vr[0]  = o0[i];
    vr[16] = o1[i];
    vr[32] = o2[i];
    vr[48] = o3[i];
  }
}

extern "C" void kernel_launch(void* const* d_in, const int* in_sizes, int n_in,
                              void* d_out, int out_size, void* d_ws, size_t ws_size,
                              hipStream_t stream) {
  (void)in_sizes; (void)n_in; (void)out_size; (void)d_ws; (void)ws_size;
  const float* qkv = (const float*)d_in[0];   // (B, L, 3, H, E) fp32
  // d_in[1] is the causal mask; causality is computed analytically instead.
  float* vout = (float*)d_out;                                  // (B,L,H,E)
  float* aout = vout + (size_t)B_ * L_ * H_ * E_;               // (B,H,L,L)
  dim3 grid(B_ * H_ * (L_ / ROWS_WG));        // 1024 workgroups
  attn_fwd_cdna5<<<grid, 256, 0, stream>>>(qkv, vout, aout);
}
